// SNN_VQVAE_COND_29772713296280
// MI455X (gfx1250) — matrix operand order repaired and checked by reading, not compiled, not run on hardware
//
#include <hip/hip_runtime.h>
#include <hip/hip_bf16.h>

typedef _Float16 h16;
typedef __attribute__((ext_vector_type(16))) _Float16 v16h;
typedef __attribute__((ext_vector_type(8)))  float    v8f;

#define TT 16
#define BB 1024
#define NN (TT*BB)
#define CDIV(a,b) (((a)+(b)-1)/(b))

// ---------------- elementwise f32 -> f16 weight convert ----------------
__global__ void k_cvt_f16(const float* __restrict__ s, h16* __restrict__ d, int n){
  int i = blockIdx.x*blockDim.x + threadIdx.x;
  if (i < n) d[i] = (h16)s[i];
}

// ---------------- class-conditioning table: table[c][o] = W[o,:] . embed[c,:] + b[o] ----
__global__ void k_class_table(const float* __restrict__ embed, const float* __restrict__ pw,
                              const float* __restrict__ pb, float* __restrict__ table,
                              int ncls, int outdim){
  int i = blockIdx.x*blockDim.x + threadIdx.x;
  if (i >= ncls*outdim) return;
  int c = i / outdim, o = i - c*outdim;
  const float* e = embed + c*128;
  const float* w = pw + o*128;
  float s = pb[o];
  for (int j = 0; j < 128; ++j) s += w[j]*e[j];
  table[i] = s;
}

// ---------------- base conv: 1->64 ch, 3x3 pad1 on 4x4 (K=9, scalar) ----------------
__global__ void k_base_conv(const float* __restrict__ x, const float* __restrict__ w,
                            h16* __restrict__ y){
  int i = blockIdx.x*blockDim.x + threadIdx.x;      // over NN*64*16
  if (i >= NN*64*16) return;
  int p = i & 15; int r = i >> 4; int co = r & 63; int n = r >> 6;
  int ho = p >> 2, wo = p & 3;
  const float* xi = x + n*16;
  const float* wk = w + co*9;
  float s = 0.f;
  for (int kh = 0; kh < 3; ++kh){
    int hi = ho + kh - 1; if (hi < 0 || hi > 3) continue;
    for (int kw = 0; kw < 3; ++kw){
      int wi = wo + kw - 1; if (wi < 0 || wi > 3) continue;
      s += xi[hi*4 + wi] * wk[kh*3 + kw];
    }
  }
  y[i] = (h16)s;
}

// ---------------- BN stats: one block per channel; writes scale/bias ----------------
__global__ __launch_bounds__(1024)
void k_bnstats(const h16* __restrict__ a, float* __restrict__ bnp, int C, int pshift){
  const int c = blockIdx.x;
  const int P = 1 << pshift;
  const int NP = NN << pshift;
  float s = 0.f, q = 0.f;
  for (int i = threadIdx.x; i < NP; i += 1024){
    int n = i >> pshift;
    int p = i & (P - 1);
    float v = (float)a[((n*C + c) << pshift) + p];
    s += v; q += v*v;
  }
  __shared__ float ss[1024], sq[1024];
  ss[threadIdx.x] = s; sq[threadIdx.x] = q;
  __syncthreads();
  for (int st = 512; st > 0; st >>= 1){
    if (threadIdx.x < st){ ss[threadIdx.x] += ss[threadIdx.x+st]; sq[threadIdx.x] += sq[threadIdx.x+st]; }
    __syncthreads();
  }
  if (threadIdx.x == 0){
    float inv = 1.f / (float)NP;
    float m = ss[0]*inv;
    float var = sq[0]*inv - m*m;
    float sc = rsqrtf(var + 1e-5f);
    bnp[c] = sc; bnp[256 + c] = -m*sc;
  }
}

// ---------------- LIF over time (tau=2, vth=1, hard reset), BN applied on the fly ----
__global__ void k_lif(const h16* __restrict__ y, const float* __restrict__ bnp,
                      h16* __restrict__ sout, int C, int pshift){
  const int CP = C << pshift;
  int i = blockIdx.x*blockDim.x + threadIdx.x;      // over BB*CP
  if (i >= BB*CP) return;
  int b = i / CP;
  int cp = i - b*CP;
  int c = cp >> pshift;
  float sc = bnp[c], bi = bnp[256 + c];
  float v = 0.f;
  for (int t = 0; t < TT; ++t){
    int id = (t*BB + b)*CP + cp;
    float xx = sc*(float)y[id] + bi;
    v += (xx - v)*0.5f;
    float spk = (v >= 1.f) ? 1.f : 0.f;
    sout[id] = (h16)spk;
    v -= spk*v;
  }
}

// stage-1 variant: spike + class conditioning (enc table row = 1024 = C*P)
__global__ void k_lif_cond(const h16* __restrict__ y, const float* __restrict__ bnp,
                           const float* __restrict__ tab, const int* __restrict__ idx,
                           h16* __restrict__ xout, int C, int pshift){
  const int CP = C << pshift;
  int i = blockIdx.x*blockDim.x + threadIdx.x;
  if (i >= BB*CP) return;
  int b = i / CP;
  int cp = i - b*CP;
  int c = cp >> pshift;
  float sc = bnp[c], bi = bnp[256 + c];
  float v = 0.f;
  for (int t = 0; t < TT; ++t){
    int id = (t*BB + b)*CP + cp;
    float xx = sc*(float)y[id] + bi;
    v += (xx - v)*0.5f;
    float spk = (v >= 1.f) ? 1.f : 0.f;
    int cls = idx[t*BB + b];
    xout[id] = (h16)(spk + tab[cls*1024 + cp]);
    v -= spk*v;
  }
}

// ---------------- residual + relu: O = relu( res + bn(Z) ); res = X or bn(X) ----------------
__global__ void k_res_relu(const h16* __restrict__ Z, const float* __restrict__ bnz,
                           const h16* __restrict__ X, const float* __restrict__ bnx,
                           h16* __restrict__ O, int C, int pshift, int total){
  int i = blockIdx.x*blockDim.x + threadIdx.x;
  if (i >= total) return;
  int c = (i >> pshift) & (C - 1);
  float z = bnz[c]*(float)Z[i] + bnz[256 + c];
  float r = (float)X[i];
  if (bnx) r = bnx[c]*r + bnx[256 + c];
  float o = r + z;
  O[i] = (h16)(o > 0.f ? o : 0.f);
}

// ---------------- WMMA implicit-GEMM conv with LDS-staged activations ----------------
// act: [N, CIN, HIN*WIN] f16, out: [N, COUT, HOUT*WOUT] f16.
// Block = (COUT/16) waves; blockIdx.x = position-tile group (PT*16 positions = NIMG images,
// contiguous in memory). All waves cooperatively stage the NIMG images into LDS with b128
// copies, then each wave computes a 16-oc x (PT*16)-position tile via v_wmma_f32_16x16x32_f16.
// Fragment lane layouts (A/B use this kernel's contiguous k order, same in both; C/D per ISA):
//   A (16x32): M = lane&15; elem e -> K = klo + e + 16*(lane>=16)
//   B (32x16): N = lane&15; elem e -> K = klo + e + 16*(lane>=16)
//   C/D (f32): elem j -> M = j + 8*(lane>=16), N = lane&15
template<int CIN, int KH, int KW, int HIN, int WIN, int HOUT, int WOUT, int STRIDE, int PAD, int PT>
__global__ __launch_bounds__(512)
void conv_wmma(const h16* __restrict__ act, const h16* __restrict__ wt,
               h16* __restrict__ out, int COUT, int NQ,
               int woc_stride, int wci_stride, int wbase){
  constexpr int POUT = HOUT*WOUT;
  constexpr int PIN  = HIN*WIN;
  constexpr int KKW  = KH*KW;
  constexpr int K    = CIN*KKW;
  constexpr int NIMG = PT*16/POUT;          // images covered by this block (contiguous)
  constexpr int TILEH = NIMG*CIN*PIN;       // halfs staged in LDS
  __shared__ __align__(32) h16 tile[TILEH];
  (void)NQ;

  const int tid = threadIdx.x;
  const int lane = tid & 31;
  const int wave = tid >> 5;
  const int laneHi = lane >> 4;
  const int lcol = lane & 15;
  const int q_base = blockIdx.x*(PT*16);
  const int n0 = q_base / POUT;             // first image of this block
  const int oc_base = wave << 4;

  // cooperative stage: NIMG contiguous images -> LDS (16B vector copies)
  {
    const uint4* __restrict__ src = (const uint4*)(act + (size_t)n0*(CIN*PIN));
    uint4* dst = (uint4*)tile;
    for (int i = tid; i < TILEH/8; i += blockDim.x) dst[i] = src[i];
  }
  __syncthreads();

  v8f acc[PT];
  const v8f zv = {0.f,0.f,0.f,0.f,0.f,0.f,0.f,0.f};
#pragma unroll
  for (int i = 0; i < PT; ++i) acc[i] = zv;

  int imgb[PT], pho[PT], pwo[PT];
#pragma unroll
  for (int pt = 0; pt < PT; ++pt){
    int q = q_base + pt*16 + lcol;
    int n = q / POUT;
    int p = q - n*POUT;
    imgb[pt] = (n - n0)*(CIN*PIN);
    pho[pt] = p / WOUT;
    pwo[pt] = p - (p/WOUT)*WOUT;
  }
  const int m = oc_base + lcol;             // A-matrix row (output channel)

  for (int klo = 0; klo < K; klo += 32){
    v16h a;
#pragma unroll
    for (int e = 0; e < 16; ++e){
      int kl = e + (laneHi << 4);
      int k = klo + kl;
      h16 val = (h16)0.f;
      if (k < K){
        int ci = k / KKW;
        int r = k - ci*KKW;
        int kh = r / KW;
        int kw = r - kh*KW;
        val = wt[m*woc_stride + ci*wci_stride + wbase + kh*3 + kw];
      }
      a[e] = val;
    }
#pragma unroll
    for (int pt = 0; pt < PT; ++pt){
      v16h b;
      if constexpr (KKW == 1 && PIN == 1 && STRIDE == 1 && PAD == 0){
        // center-tap / 1x1 on 1x1 input: K-dim contiguous in LDS -> wide ds loads
        b = *reinterpret_cast<const v16h*>(tile + imgb[pt] + klo + (laneHi << 4));
      } else {
#pragma unroll
        for (int e = 0; e < 16; ++e){
          int k = klo + e + (laneHi << 4);
          h16 val = (h16)0.f;
          if (k < K){
            int ci = k / KKW;
            int r = k - ci*KKW;
            int kh = r / KW;
            int kw = r - kh*KW;
            int hi = pho[pt]*STRIDE + kh - PAD;
            int wi = pwo[pt]*STRIDE + kw - PAD;
            if (hi >= 0 && hi < HIN && wi >= 0 && wi < WIN)
              val = tile[imgb[pt] + ci*PIN + hi*WIN + wi];
          }
          b[e] = val;
        }
      }
      acc[pt] = __builtin_amdgcn_wmma_f32_16x16x32_f16(false, a, false, b,
                                                       (short)0, acc[pt], false, false);
    }
  }
#pragma unroll
  for (int pt = 0; pt < PT; ++pt){
    int q = q_base + pt*16 + lcol;
    int n = q / POUT;
    int p = q - n*POUT;
#pragma unroll
    for (int j = 0; j < 8; ++j){
      int oc = oc_base + j + (laneHi << 3);
      out[(n*COUT + oc)*POUT + p] = (h16)acc[pt][j];
    }
  }
}

// ---------------- decoder ----------------
__global__ void k_dec_h0(const h16* __restrict__ z, const float* __restrict__ tab,
                         const int* __restrict__ idx, float* __restrict__ h0){
  int i = blockIdx.x*blockDim.x + threadIdx.x;      // NN*256
  if (i >= NN*256) return;
  int n = i >> 8, c = i & 255;
  h0[i] = (float)z[i] + tab[idx[n]*256 + c];
}

__global__ void k_mm(const float* __restrict__ in, const float* __restrict__ w,
                     const float* __restrict__ b, float* __restrict__ out, int I, int O){
  int t = blockIdx.x*blockDim.x + threadIdx.x;      // NN*O
  if (t >= NN*O) return;
  int n = t / O, o = t - n*O;
  const float* xi = in + n*I;
  float s = b[o];
  for (int j = 0; j < I; ++j) s += xi[j]*w[j*O + o];
  out[t] = s;
}

// ConvTranspose2d 64->32, k3 s2 p1 op1 on 1x1 -> 2x2: out[co,oh,ow] = b + sum_ci x[ci]*w[ci,co,oh+1,ow+1]
__global__ void k_dc4(const float* __restrict__ h, const float* __restrict__ w,
                      const float* __restrict__ b, float* __restrict__ out){
  int i = blockIdx.x*blockDim.x + threadIdx.x;      // NN*32*4
  if (i >= NN*32*4) return;
  int p = i & 3; int r = i >> 2; int co = r & 31; int n = r >> 5;
  int kh = (p >> 1) + 1, kw = (p & 1) + 1;
  const float* x = h + n*64;
  float s = b[co];
  for (int ci = 0; ci < 64; ++ci)
    s += x[ci]*w[((ci*32 + co)*3 + kh)*3 + kw];
  out[i] = s;
}

// fused dc5 (32->1, k3 s2 p1 op1, 2x2 -> 4x4) + membrane sum over t + tanh + MSE partials
__global__ __launch_bounds__(256)
void k_loss(const float* __restrict__ h4, const float* __restrict__ w5,
            const float* __restrict__ b5, const float* __restrict__ image,
            float* __restrict__ part){
  int i = blockIdx.x*256 + threadIdx.x;             // BB*16
  float sqd = 0.f;
  if (i < BB*16){
    int p = i & 15; int b = i >> 4;
    int oh = p >> 2, ow = p & 3;
    int ih = (oh >= 2) ? 1 : 0;  int kh = oh + 1 - 2*ih;
    int iw = (ow >= 2) ? 1 : 0;  int kw = ow + 1 - 2*iw;
    float m = 0.f, cf = 1.f;
    for (int t = TT - 1; t >= 0; --t){
      const float* x = h4 + (t*BB + b)*128;         // [32,2,2]
      float s = b5[0];
      for (int ci = 0; ci < 32; ++ci)
        s += x[ci*4 + ih*2 + iw]*w5[ci*9 + kh*3 + kw];
      m += cf*s;
      cf *= 0.8f;
    }
    float d = tanhf(m) - image[i];
    sqd = d*d;
  }
  __shared__ float red[256];
  red[threadIdx.x] = sqd; __syncthreads();
  for (int st = 128; st > 0; st >>= 1){
    if (threadIdx.x < st) red[threadIdx.x] += red[threadIdx.x + st];
    __syncthreads();
  }
  if (threadIdx.x == 0) part[blockIdx.x] = red[0];
}

__global__ void k_final(const float* __restrict__ part, float* __restrict__ out){
  float s = 0.f;
  for (int i = 0; i < 64; ++i) s += part[i];
  float m = s / (float)(BB*16);
  out[0] = 0.f; out[1] = m; out[2] = m;  // (0, recon_loss, real_recon_loss); DATA_VARIANCE=1
}

// ================= host launch =================
extern "C" void kernel_launch(void* const* d_in, const int* in_sizes, int n_in,
                              void* d_out, int out_size, void* d_ws, size_t ws_size,
                              hipStream_t stream){
  (void)in_sizes; (void)n_in; (void)out_size; (void)ws_size;
  const float* x         = (const float*)d_in[0];
  const float* image     = (const float*)d_in[1];
  const int*   idx       = (const int*)  d_in[2];
  const float* enc_embed = (const float*)d_in[3];
  const float* enc_pw    = (const float*)d_in[4];
  const float* enc_pb    = (const float*)d_in[5];
  const float* base_w    = (const float*)d_in[6];

  char* ws = (char*)d_ws;
  size_t off = 0;
  auto carve = [&](size_t bytes)->void*{
    void* p = ws + off; off += (bytes + 255) & ~(size_t)255; return p;
  };
  h16* actX = (h16*)carve((size_t)NN*64*16*2);   // block input / output
  h16* actY = (h16*)carve((size_t)NN*64*16*2);   // conv1 out
  h16* actS = (h16*)carve((size_t)NN*64*16*2);   // spikes
  h16* actZ = (h16*)carve((size_t)NN*64*16*2);   // conv2 out
  h16* actD = (h16*)carve((size_t)NN*128*4*2);   // downsample out
  float* enc_table = (float*)carve(46*1024*4);
  float* dec_table = (float*)carve(46*256*4);
  float* bnp0 = (float*)carve(512*4);
  float* bnp1 = (float*)carve(512*4);
  float* bnp2 = (float*)carve(512*4);
  float* dec0 = (float*)carve((size_t)NN*256*4);
  float* dec1 = (float*)carve((size_t)NN*128*4);
  float* dec2 = (float*)carve((size_t)NN*64*4);
  float* h4   = (float*)carve((size_t)NN*128*4);
  float* part = (float*)carve(64*4);

  // f16 conv weights (order matches d_in[7..20])
  const int wsz[14] = {36864,36864,36864,36864,          // l1b1_w1/w2, l1b2_w1/w2
                       73728,147456,8192,                // l2b1_w1/w2/wd
                       147456,147456,                    // l2b2_w1/w2
                       294912,589824,32768,              // l3b1_w1/w2/wd
                       589824,589824};                   // l3b2_w1/w2
  h16* wh[14];
  for (int i = 0; i < 14; ++i){
    wh[i] = (h16*)carve((size_t)wsz[i]*2);
    k_cvt_f16<<<CDIV(wsz[i],256),256,0,stream>>>((const float*)d_in[7+i], wh[i], wsz[i]);
  }

  // conditioning tables
  k_class_table<<<CDIV(46*1024,256),256,0,stream>>>(enc_embed, enc_pw, enc_pb, enc_table, 46, 1024);
  k_class_table<<<CDIV(46*256,256),256,0,stream>>>((const float*)d_in[21], (const float*)d_in[22],
                                                   (const float*)d_in[23], dec_table, 46, 256);

  // ---- stage 1: base conv -> bn -> lif -> +cond ----
  k_base_conv<<<CDIV(NN*64*16,256),256,0,stream>>>(x, base_w, actY);
  k_bnstats<<<64,1024,0,stream>>>(actY, bnp0, 64, 4);
  k_lif_cond<<<CDIV(BB*64*16,256),256,0,stream>>>(actY, bnp0, enc_table, idx, actX, 64, 4);

  // ---- layer1: two identity resblocks, 64ch @ 4x4 (4 waves/block, 8KB LDS) ----
  for (int blk = 0; blk < 2; ++blk){
    h16* w1 = wh[blk*2 + 0];
    h16* w2 = wh[blk*2 + 1];
    conv_wmma<64,3,3,4,4,4,4,1,1,4><<<NN*16/64,128,0,stream>>>(actX, w1, actY, 64, NN*16, 576, 9, 0);
    k_bnstats<<<64,1024,0,stream>>>(actY, bnp0, 64, 4);
    k_lif<<<CDIV(BB*64*16,256),256,0,stream>>>(actY, bnp0, actS, 64, 4);
    conv_wmma<64,3,3,4,4,4,4,1,1,4><<<NN*16/64,128,0,stream>>>(actS, w2, actZ, 64, NN*16, 576, 9, 0);
    k_bnstats<<<64,1024,0,stream>>>(actZ, bnp1, 64, 4);
    k_res_relu<<<CDIV(NN*64*16,256),256,0,stream>>>(actZ, bnp1, actX, (const float*)nullptr,
                                                    actX, 64, 4, NN*64*16);
  }

  // ---- layer2 block1: 64->128, stride 2 (4x4 -> 2x2), with 1x1 downsample ----
  conv_wmma<64,3,3,4,4,2,2,2,1,2><<<NN*4/32,256,0,stream>>>(actX, wh[4], actY, 128, NN*4, 576, 9, 0);
  k_bnstats<<<128,1024,0,stream>>>(actY, bnp0, 128, 2);
  k_lif<<<CDIV(BB*128*4,256),256,0,stream>>>(actY, bnp0, actS, 128, 2);
  conv_wmma<128,3,3,2,2,2,2,1,1,4><<<NN*4/64,256,0,stream>>>(actS, wh[5], actZ, 128, NN*4, 1152, 9, 0);
  k_bnstats<<<128,1024,0,stream>>>(actZ, bnp1, 128, 2);
  conv_wmma<64,1,1,4,4,2,2,2,0,2><<<NN*4/32,256,0,stream>>>(actX, wh[6], actD, 128, NN*4, 64, 1, 0);
  k_bnstats<<<128,1024,0,stream>>>(actD, bnp2, 128, 2);
  k_res_relu<<<CDIV(NN*128*4,256),256,0,stream>>>(actZ, bnp1, actD, bnp2, actX, 128, 2, NN*128*4);

  // ---- layer2 block2: identity, 128ch @ 2x2 ----
  conv_wmma<128,3,3,2,2,2,2,1,1,4><<<NN*4/64,256,0,stream>>>(actX, wh[7], actY, 128, NN*4, 1152, 9, 0);
  k_bnstats<<<128,1024,0,stream>>>(actY, bnp0, 128, 2);
  k_lif<<<CDIV(BB*128*4,256),256,0,stream>>>(actY, bnp0, actS, 128, 2);
  conv_wmma<128,3,3,2,2,2,2,1,1,4><<<NN*4/64,256,0,stream>>>(actS, wh[8], actZ, 128, NN*4, 1152, 9, 0);
  k_bnstats<<<128,1024,0,stream>>>(actZ, bnp1, 128, 2);
  k_res_relu<<<CDIV(NN*128*4,256),256,0,stream>>>(actZ, bnp1, actX, (const float*)nullptr,
                                                  actX, 128, 2, NN*128*4);

  // ---- layer3 block1: 128->256, stride 2 (2x2 -> 1x1), downsample; conv2 = center-tap 1x1 ----
  conv_wmma<128,3,3,2,2,1,1,2,1,2><<<NN/32,512,0,stream>>>(actX, wh[9], actY, 256, NN, 1152, 9, 0);
  k_bnstats<<<256,1024,0,stream>>>(actY, bnp0, 256, 0);
  k_lif<<<CDIV(BB*256,256),256,0,stream>>>(actY, bnp0, actS, 256, 0);
  conv_wmma<256,1,1,1,1,1,1,1,0,2><<<NN/32,512,0,stream>>>(actS, wh[10], actZ, 256, NN, 2304, 9, 4);
  k_bnstats<<<256,1024,0,stream>>>(actZ, bnp1, 256, 0);
  conv_wmma<128,1,1,2,2,1,1,2,0,2><<<NN/32,512,0,stream>>>(actX, wh[11], actD, 256, NN, 128, 1, 0);
  k_bnstats<<<256,1024,0,stream>>>(actD, bnp2, 256, 0);
  k_res_relu<<<CDIV(NN*256,256),256,0,stream>>>(actZ, bnp1, actD, bnp2, actX, 256, 0, NN*256);

  // ---- layer3 block2: identity; both convs are center-tap 1x1 (input 1x1, pad 1) ----
  conv_wmma<256,1,1,1,1,1,1,1,0,2><<<NN/32,512,0,stream>>>(actX, wh[12], actY, 256, NN, 2304, 9, 4);
  k_bnstats<<<256,1024,0,stream>>>(actY, bnp0, 256, 0);
  k_lif<<<CDIV(BB*256,256),256,0,stream>>>(actY, bnp0, actS, 256, 0);
  conv_wmma<256,1,1,1,1,1,1,1,0,2><<<NN/32,512,0,stream>>>(actS, wh[13], actZ, 256, NN, 2304, 9, 4);
  k_bnstats<<<256,1024,0,stream>>>(actZ, bnp1, 256, 0);
  k_res_relu<<<CDIV(NN*256,256),256,0,stream>>>(actZ, bnp1, actX, (const float*)nullptr,
                                                actX, 256, 0, NN*256);

  // ---- decoder ----
  k_dec_h0<<<CDIV(NN*256,256),256,0,stream>>>(actX, dec_table, idx, dec0);
  k_mm<<<CDIV(NN*128,256),256,0,stream>>>(dec0, (const float*)d_in[24], (const float*)d_in[25], dec1, 256, 128);
  k_mm<<<CDIV(NN*64,256),256,0,stream>>>(dec1, (const float*)d_in[26], (const float*)d_in[27], dec2, 128, 64);
  k_dc4<<<CDIV(NN*32*4,256),256,0,stream>>>(dec2, (const float*)d_in[28], (const float*)d_in[29], h4);
  k_loss<<<64,256,0,stream>>>(h4, (const float*)d_in[30], (const float*)d_in[31], image, part);
  k_final<<<1,1,0,stream>>>(part, (float*)d_out);
}
